// Net_44822278701438
// MI455X (gfx1250) — matrix operand order repaired and checked by reading, not compiled, not run on hardware
//
#include <hip/hip_runtime.h>

// ---------- types ----------
typedef __attribute__((ext_vector_type(16))) _Float16 v16h;
typedef __attribute__((ext_vector_type(4)))  _Float16 v4h;
typedef __attribute__((ext_vector_type(8)))  float    v8f;
typedef __attribute__((ext_vector_type(4)))  float    vf4;

#define KDIM 256          // K of every dense GEMM in this net
#define KP   264          // LDS row stride in halves (pad 8 -> conflict-free b128)

union Frag16 {            // one WMMA 16-bit operand: 16 halves = 8 VGPRs
  v16h h;
  vf4  q[2];
};

// ---------- elementwise kernels ----------
__global__ void cast_f32_f16(const float* __restrict__ src,
                             unsigned short* __restrict__ dst, long total4) {
  long i = (long)blockIdx.x * blockDim.x + threadIdx.x;
  if (i >= total4) return;
  vf4 v = ((const vf4*)src)[i];
  v4h h;
#pragma unroll
  for (int j = 0; j < 4; ++j) h[j] = (_Float16)v[j];
  ((v4h*)dst)[i] = h;
}

// W[k][n] (f32, K x Nn) -> Wt[n][k] (f16, Nn x K)
__global__ void wcast_transpose(const float* __restrict__ W,
                                unsigned short* __restrict__ Wt, int K, int Nn) {
  int i = blockIdx.x * blockDim.x + threadIdx.x;
  if (i >= K * Nn) return;
  int k = i % K, n = i / K;                 // writes coalesced along k
  reinterpret_cast<_Float16*>(Wt)[(size_t)n * K + k] = (_Float16)W[(size_t)k * Nn + n];
}

__global__ void fill_zero(float* __restrict__ p, long total4) {
  long i = (long)blockIdx.x * blockDim.x + threadIdx.x;
  if (i >= total4) return;
  vf4 z = {0.f, 0.f, 0.f, 0.f};
  ((vf4*)p)[i] = z;
}

__global__ void bias_relu(float* __restrict__ S, const float* __restrict__ b, long total4) {
  long i = (long)blockIdx.x * blockDim.x + threadIdx.x;
  if (i >= total4) return;
  int cchunk = (int)(i & 63);               // 256 cols = 64 float4 chunks per row
  vf4 bb = ((const vf4*)b)[cchunk];
  vf4 v = ((vf4*)S)[i];
#pragma unroll
  for (int j = 0; j < 4; ++j) v[j] = fmaxf(v[j] + bb[j], 0.f);
  ((vf4*)S)[i] = v;
}

// piecewise multi-term ReLU: w1*soft(0.1) - w2*soft(0.5); w1=1.8, w2=0.8
__global__ void act_piecewise(float* __restrict__ P, long total4) {
  long i = (long)blockIdx.x * blockDim.x + threadIdx.x;
  if (i >= total4) return;
  vf4 v = ((vf4*)P)[i];
#pragma unroll
  for (int j = 0; j < 4; ++j) {
    float x  = v[j];
    float s1 = fmaxf(x - 0.1f, 0.f) - fmaxf(-x - 0.1f, 0.f);
    float s2 = fmaxf(x - 0.5f, 0.f) - fmaxf(-x - 0.5f, 0.f);
    v[j] = 1.8f * s1 - 0.8f * s2;
  }
  ((vf4*)P)[i] = v;
}

// COO SpMM: dst[row[e]] += scale*val[e]*src[col[e]]  (64 threads/edge, 4 cols each)
__global__ void spmm_atomic(const int* __restrict__ idx, const float* __restrict__ val,
                            const float* __restrict__ src, float* __restrict__ dst,
                            int E, float scale) {
  long gid = (long)blockIdx.x * blockDim.x + threadIdx.x;
  int e = (int)(gid >> 6);
  if (e >= E) return;
  int c4 = ((int)gid & 63) << 2;
  int r = idx[e];
  int c = idx[(size_t)E + e];
  float v = scale * val[e];
  vf4 s = *(const vf4*)(src + (size_t)c * 256 + c4);
  float* d = dst + (size_t)r * 256 + c4;
  atomicAdd(d + 0, v * s[0]);
  atomicAdd(d + 1, v * s[1]);
  atomicAdd(d + 2, v * s[2]);
  atomicAdd(d + 3, v * s[3]);
}

// ---------- WMMA GEMM ----------
// C[N x OUTN] (+)= A16[N x 256] @ W,  W given transposed+f16: Wt[OUTN x 256].
// Block: 256 threads = 8 waves, 128 rows; blockIdx.y covers NTB-column slabs.
// EPI: 0 = store, 1 = accumulate, 2 = tanh(acc + bias) store.
template <int OUTN, int NTB, int EPI>
__global__ __launch_bounds__(256) void gemm_wmma(
    const unsigned short* __restrict__ A16p,
    const unsigned short* __restrict__ Wtp,
    float* __restrict__ C,
    const float* __restrict__ bias,
    int N) {
  extern __shared__ unsigned short sB[];     // [NTB][KP] halves
  const _Float16* A16 = reinterpret_cast<const _Float16*>(A16p);
  constexpr int NT = NTB / 16;

  const int tid   = threadIdx.x;
  const int nbase = blockIdx.y * NTB;

  // Stage the transposed f16 weight slab into LDS with the CDNA5 async path:
  // per-lane 16B global -> LDS copies, tracked by ASYNCcnt (bypasses VGPRs).
  {
    const unsigned sBase = (unsigned)(uintptr_t)(void*)&sB[0];  // LDS byte offset
    const int chunks = NTB * (KDIM / 8);      // 16-byte (8-half) chunks
    for (int i = tid; i < chunks; i += 256) {
      int n  = i / (KDIM / 8);
      int k8 = (i % (KDIM / 8)) * 8;
      unsigned lds_off = sBase + (unsigned)(n * KP + k8) * 2u;
      const unsigned short* g = Wtp + (size_t)(nbase + n) * KDIM + k8;
      asm volatile("global_load_async_to_lds_b128 %0, %1, off"
                   :: "v"(lds_off), "v"(g)
                   : "memory");
    }
    asm volatile("s_wait_asynccnt 0" ::: "memory");
  }
  __syncthreads();

  const int wave  = tid >> 5;
  const int lane  = tid & 31;
  const int mrow  = lane & 15;
  const int khalf = lane >> 4;
  const int mbase = blockIdx.x * 128 + wave * 16;

  int mA = mbase + mrow;
  if (mA > N - 1) mA = N - 1;                // clamp: loads stay in-bounds
  const _Float16* aRow = A16 + (size_t)mA * KDIM;

  // LDS byte base for this lane's B fragments: row = tile*16 + (lane&15)
  const int nlane   = lane & 15;
  const int bKhalf  = khalf * 8;             // B 32x16: ISA table K order (matches A)

  v8f acc[NT] = {};

#pragma unroll
  for (int kt = 0; kt < KDIM / 32; ++kt) {
    Frag16 a;
    const int k0a = kt * 32 + khalf * 8;     // ISA A 16x32 f16 layout
    a.q[0] = *(const vf4*)(aRow + k0a);
    a.q[1] = *(const vf4*)(aRow + k0a + 16);

    const int k0b = kt * 32 + bKhalf;
    Frag16 b[2];                             // double-buffered B fragments
    b[0].q[0] = *(const vf4*)(&sB[nlane * KP + k0b]);
    b[0].q[1] = *(const vf4*)(&sB[nlane * KP + k0b + 16]);
#pragma unroll
    for (int t = 0; t < NT; ++t) {
      if (t + 1 < NT) {
        const int n1 = (t + 1) * 16 + nlane;
        b[(t + 1) & 1].q[0] = *(const vf4*)(&sB[n1 * KP + k0b]);
        b[(t + 1) & 1].q[1] = *(const vf4*)(&sB[n1 * KP + k0b + 16]);
      }
      acc[t] = __builtin_amdgcn_wmma_f32_16x16x32_f16(
          false, a.h, false, b[t & 1].h, (short)0, acc[t], false, false);
    }
  }

  // epilogue: VGPR r -> row mbase + khalf*8 + r, col = lane&15 within tile
  const int colLane = lane & 15;
  const int rOff    = khalf * 8;
#pragma unroll
  for (int t = 0; t < NT; ++t) {
#pragma unroll
    for (int r = 0; r < 8; ++r) {
      int row = mbase + rOff + r;
      if (row < N) {
        int col = nbase + t * 16 + colLane;
        size_t off = (size_t)row * OUTN + col;
        float v = acc[t][r];
        if (EPI == 0)      C[off] = v;
        else if (EPI == 1) C[off] += v;
        else               C[off] = tanhf(v + bias[col]);
      }
    }
  }
}

// ---------- host ----------
static inline long up256(long x) { return (x + 255) & ~255L; }

extern "C" void kernel_launch(void* const* d_in, const int* in_sizes, int n_in,
                              void* d_out, int out_size, void* d_ws, size_t ws_size,
                              hipStream_t stream) {
  const float* x     = (const float*)d_in[0];
  const int*   A_idx = (const int*)d_in[1];
  const float* A_val = (const float*)d_in[2];
  const int*   L_idx = (const int*)d_in[3];
  const float* L_val = (const float*)d_in[4];
  const int N  = in_sizes[0] / 256;
  const int EA = in_sizes[2];
  const int EL = in_sizes[4];

  const float* gcn_w[2] = {(const float*)d_in[5],  (const float*)d_in[13]};
  const float* gcn_b[2] = {(const float*)d_in[6],  (const float*)d_in[14]};
  const float* pai1[2]  = {(const float*)d_in[7],  (const float*)d_in[15]};
  const float* pai2[2]  = {(const float*)d_in[8],  (const float*)d_in[16]};
  const float* w1w[2]   = {(const float*)d_in[9],  (const float*)d_in[17]};
  const float* w1b[2]   = {(const float*)d_in[10], (const float*)d_in[18]};
  const float* w2w[2]   = {(const float*)d_in[11], (const float*)d_in[19]};
  const float* w2b[2]   = {(const float*)d_in[12], (const float*)d_in[20]};

  // workspace layout
  char* ws = (char*)d_ws;
  const size_t szH = up256((size_t)N * 256 * 2);
  const size_t szF = up256((size_t)N * 256 * 4);
  unsigned short* X16 = (unsigned short*)ws;        ws += szH;  // f16(x)
  unsigned short* T16 = (unsigned short*)ws;        ws += szH;  // f16(current Z)
  float* bufA = (float*)ws;                         ws += szF;  // Z @ gcn_w
  float* bufB = (float*)ws;                         ws += szF;  // SpMM accum -> Z_tp
  float* bufC = (float*)ws;                         ws += szF;  // embed pre-act -> Z_embed
  unsigned short* W16 = (unsigned short*)ws;        ws += up256(256 * 256 * 2);
  (void)ws_size; (void)n_in; (void)out_size;

  float* out = (float*)d_out;
  const size_t n64 = (size_t)N * 64;

  const long ew4   = (long)N * 64;                  // N*256 / 4
  const int  EWG   = (int)((ew4 + 255) / 256);
  const int  GX    = (N + 127) / 128;
  const dim3 gBig(GX, 2), gSmall(GX, 1);
  const size_t shBig = 128 * KP * 2, shSmall = 64 * KP * 2;
  const int spA = (int)(((long)EA * 64 + 255) / 256);
  const int spL = (int)(((long)EL * 64 + 255) / 256);

  auto wcast = [&](const float* W, int K, int Nn) {
    wcast_transpose<<<(K * Nn + 255) / 256, 256, 0, stream>>>(W, W16, K, Nn);
  };

  cast_f32_f16<<<EWG, 256, 0, stream>>>(x, X16, ew4);

  const unsigned short* Zin16 = X16;                 // layer-0 input = x
  for (int i = 0; i < 2; ++i) {
    // GCN: bufB = relu(A @ (Z @ gcn_w) + b)
    wcast(gcn_w[i], 256, 256);
    gemm_wmma<256, 128, 0><<<gBig, 256, shBig, stream>>>(Zin16, W16, bufA, nullptr, N);
    fill_zero<<<EWG, 256, 0, stream>>>(bufB, ew4);
    spmm_atomic<<<spA, 256, 0, stream>>>(A_idx, A_val, bufA, bufB, EA, 1.0f);
    bias_relu<<<EWG, 256, 0, stream>>>(bufB, gcn_b[i], ew4);
    cast_f32_f16<<<EWG, 256, 0, stream>>>(bufB, T16, ew4);

    // z_tp output: tanh(Z_tp @ w1_w + b)
    wcast(w1w[i], 256, 64);
    gemm_wmma<64, 64, 2><<<gSmall, 256, shSmall, stream>>>(T16, W16, out + (size_t)i * n64, w1b[i], N);

    // embed: bufC = Z_tp@pai1 + X@pai2 - lamda * L@Z_tp, then activation
    wcast(pai1[i], 256, 256);
    gemm_wmma<256, 128, 0><<<gBig, 256, shBig, stream>>>(T16, W16, bufC, nullptr, N);
    wcast(pai2[i], 256, 256);
    gemm_wmma<256, 128, 1><<<gBig, 256, shBig, stream>>>(X16, W16, bufC, nullptr, N);
    spmm_atomic<<<spL, 256, 0, stream>>>(L_idx, L_val, bufB, bufC, EL, -1.0f);  // LAMDA=1
    act_piecewise<<<EWG, 256, 0, stream>>>(bufC, ew4);
    cast_f32_f16<<<EWG, 256, 0, stream>>>(bufC, T16, ew4);

    // z_embed output: tanh(Z_embed @ w2_w + b)
    wcast(w2w[i], 256, 64);
    gemm_wmma<64, 64, 2><<<gSmall, 256, shSmall, stream>>>(T16, W16, out + (size_t)(2 + i) * n64, w2b[i], N);

    Zin16 = T16;                                    // next layer's Z_tp = Z_embed
  }
}
